// DBA_57956288692357
// MI455X (gfx1250) — matrix operand order repaired and checked by reading, not compiled, not run on hardware
//
#include <hip/hip_runtime.h>
#include <hip/hip_bf16.h>
#include <math.h>

typedef __attribute__((ext_vector_type(16))) _Float16 v16h;
typedef __attribute__((ext_vector_type(8)))  _Float16 v8h;
typedef __attribute__((ext_vector_type(8)))  float    v8f;

#define TPB 256
#define FINP_MAX 160
#define BS_STRIDE 24   // LDS row stride in halfs (48B): 16B-aligned, conflict-spreading

// ---------- helpers ----------
__device__ __forceinline__ unsigned int ord_enc(float f) {
  unsigned int u = __float_as_uint(f);
  return (u & 0x80000000u) ? ~u : (u | 0x80000000u);
}
__device__ __forceinline__ float ord_dec(unsigned int u) {
  return __uint_as_float((u & 0x80000000u) ? (u & 0x7FFFFFFFu) : ~u);
}

// ---------- fills ----------
__global__ void k_fill_u32(unsigned int* p, unsigned int v, int cnt) {
  int i = blockIdx.x * blockDim.x + threadIdx.x;
  if (i < cnt) p[i] = v;
}
__global__ void k_fill_u8(unsigned char* p, unsigned char v, int cnt) {
  int i = blockIdx.x * blockDim.x + threadIdx.x;
  if (i < cnt) p[i] = v;
}

// ---------- f32 -> f16 convert with zero padding ----------
__global__ void k_cvt_pad(const float* __restrict__ in, _Float16* __restrict__ out,
                          int n, int f, int nP, int fP) {
  int i = blockIdx.x * blockDim.x + threadIdx.x;
  if (i >= nP * fP) return;
  int r = i / fP, c = i - r * fP;
  out[i] = (r < n && c < f) ? (_Float16)in[(size_t)r * f + c] : (_Float16)0.0f;
}

// ---------- WMMA GEMM on padded f16:  Y[n,fout] = X @ W + bias ----------
// X: [nP, finP] f16 (nP mult16, finP mult32, zero padded)
// W: [finP, foutP] f16 (foutP mult16, zero padded)
// block = 256 (8 waves), one 16-wide column tile per block staged in LDS,
// each wave computes one 16x16 tile. Inner loop: 2x global_load_b128 (A)
// + 16x strided 2-byte LDS loads (B) + 1x v_wmma_f32_16x16x32_f16, no guards.
__global__ __launch_bounds__(256) void k_gemm_f16(
    const _Float16* __restrict__ X, const _Float16* __restrict__ W,
    const float* __restrict__ bias, float* __restrict__ Y,
    int n, int fout, int finP, int foutP) {
  __shared__ _Float16 Bs[FINP_MAX * BS_STRIDE];
  const int colTile = blockIdx.y * 16;
  // cooperative stage of W[:, colTile:colTile+16] (all K) into LDS, v8h chunks
  for (int c = threadIdx.x; c < finP * 2; c += 256) {
    int k = c >> 1, off = (c & 1) << 3;
    *(v8h*)(Bs + k * BS_STRIDE + off) =
        *(const v8h*)(W + (size_t)k * foutP + colTile + off);
  }
  __syncthreads();

  const int lane = threadIdx.x & 31;
  const int wave = threadIdx.x >> 5;
  const int rowTile = (blockIdx.x * 8 + wave) * 16;
  if (rowTile >= n) return;                 // uniform per wave (after barrier)
  const int m = lane & 15;                  // A row in tile / C col
  const int khalf = lane >> 4;              // A K-half select
  v8f c = {};
  // A: lane holds row m, halves K = k0 + khalf*8 + (0..7) and +16
  const _Float16* xp = X + (size_t)(rowTile + m) * finP + (khalf << 3);
  // B: lane holds column N = m, K = k0 + khalf*8 + (0..7) and +16 (from LDS)
  const _Float16* bp = Bs + (khalf << 3) * BS_STRIDE + m;
  for (int k0 = 0; k0 < finP; k0 += 32) {
    v8h alo = *(const v8h*)(xp);
    v8h ahi = *(const v8h*)(xp + 16);
    v16h b;
    #pragma unroll
    for (int s = 0; s < 8; ++s) { b[s] = bp[s * BS_STRIDE]; b[s + 8] = bp[(s + 16) * BS_STRIDE]; }
    xp += 32;
    bp += 32 * BS_STRIDE;
    v16h a = __builtin_shufflevector(alo, ahi, 0,1,2,3,4,5,6,7,8,9,10,11,12,13,14,15);

    c = __builtin_amdgcn_wmma_f32_16x16x32_f16(
        /*neg_a=*/false, a, /*neg_b=*/false, b,
        /*c_mod=*/(short)0, c, /*reuse_a=*/false, /*reuse_b=*/false);
  }
  // C/D: VGPR r -> M = r + 8*khalf, N = lane&15
  #pragma unroll
  for (int r = 0; r < 8; ++r) {
    int row = rowTile + r + (khalf << 3);
    int col = colTile + m;
    if (row < n && col < fout) Y[(size_t)row * fout + col] = c[r] + bias[col];
  }
}

// ---------- elementwise ----------
__global__ void k_elu(float* x, int cnt) {
  int i = blockIdx.x * blockDim.x + threadIdx.x;
  if (i < cnt) { float v = x[i]; x[i] = v > 0.0f ? v : (expf(v) - 1.0f); }
}
__global__ void k_tanh(const float* in, float* out, int n) {
  int i = blockIdx.x * blockDim.x + threadIdx.x;
  if (i < n) out[i] = tanhf(in[i]);
}
__global__ void k_concat_pos(const float* __restrict__ a, const float* __restrict__ pos,
                             float* __restrict__ out, int n, int Fa) {
  int i = blockIdx.x * blockDim.x + threadIdx.x;
  int Ft = Fa + 3;
  if (i >= n * Ft) return;
  int row = i / Ft, c = i - row * Ft;
  out[i] = (c < Fa) ? a[(size_t)row * Fa + c] : pos[row * 3 + (c - Fa)];
}
__global__ void k_eattr(const int* __restrict__ src, const int* __restrict__ dst,
                        const float* __restrict__ pos, float* __restrict__ ea, int E) {
  int e = blockIdx.x * blockDim.x + threadIdx.x;
  if (e >= E) return;
  int s = src[e], d = dst[e];
  #pragma unroll
  for (int c = 0; c < 3; ++c) ea[e * 3 + c] = pos[d * 3 + c] - pos[s * 3 + c];
}
__global__ void k_deg(const int* __restrict__ src, float* deg, int E) {
  int e = blockIdx.x * blockDim.x + threadIdx.x;
  if (e < E) atomicAdd(deg + src[e], 1.0f);
}

// ---------- GATv2 sparse plumbing ----------
__global__ void k_slattr_accum(const int* __restrict__ dst, const float* __restrict__ ea,
                               const unsigned char* __restrict__ mask, float* acc, int E) {
  int e = blockIdx.x * blockDim.x + threadIdx.x;
  if (e >= E) return;
  if (mask && !mask[e]) return;
  int d = dst[e];
  atomicAdd(acc + d * 4 + 0, ea[e * 3 + 0]);
  atomicAdd(acc + d * 4 + 1, ea[e * 3 + 1]);
  atomicAdd(acc + d * 4 + 2, ea[e * 3 + 2]);
  atomicAdd(acc + d * 4 + 3, 1.0f);
}
__global__ void k_slattr_norm(const float* acc, float* sl, int n) {
  int i = blockIdx.x * blockDim.x + threadIdx.x;
  if (i >= n) return;
  float c = fmaxf(acc[i * 4 + 3], 1.0f);
  sl[i * 3 + 0] = acc[i * 4 + 0] / c;
  sl[i * 3 + 1] = acc[i * 4 + 1] / c;
  sl[i * 3 + 2] = acc[i * 4 + 2] / c;
}
// per (edge|selfloop): s = sum_f leaky(xl[src]+xr[dst]+ea·We[:,f]) * att[f]; atomicMax m[dst]
__global__ void k_edge_score(const int* __restrict__ src, const int* __restrict__ dst,
                             const float* __restrict__ ea, const float* __restrict__ sl,
                             const unsigned char* __restrict__ mask,
                             const float* __restrict__ xl, const float* __restrict__ xr,
                             const float* __restrict__ We, const float* __restrict__ att,
                             float* __restrict__ sout, unsigned int* __restrict__ menc,
                             int E, int n, int F) {
  int e = blockIdx.x * blockDim.x + threadIdx.x;
  if (e >= E + n) return;
  int si, di; float a0, a1, a2; bool ok;
  if (e < E) {
    si = src[e]; di = dst[e];
    ok = mask ? (mask[e] != 0) : true;
    a0 = ea[e * 3]; a1 = ea[e * 3 + 1]; a2 = ea[e * 3 + 2];
  } else {
    si = di = e - E; ok = true;
    a0 = sl[si * 3]; a1 = sl[si * 3 + 1]; a2 = sl[si * 3 + 2];
  }
  float s = 0.0f;
  if ((F & 3) == 0) {
    const float4* l4 = (const float4*)(xl + (size_t)si * F);
    const float4* r4 = (const float4*)(xr + (size_t)di * F);
    for (int f = 0; f < F; f += 4) {
      float4 l = l4[f >> 2], r = r4[f >> 2];
      float4 w0 = *(const float4*)(We + f);
      float4 w1 = *(const float4*)(We + F + f);
      float4 w2 = *(const float4*)(We + 2 * F + f);
      float4 at = *(const float4*)(att + f);
      float v;
      v = l.x + r.x + a0 * w0.x + a1 * w1.x + a2 * w2.x; v = v > 0.f ? v : 0.2f * v; s += v * at.x;
      v = l.y + r.y + a0 * w0.y + a1 * w1.y + a2 * w2.y; v = v > 0.f ? v : 0.2f * v; s += v * at.y;
      v = l.z + r.z + a0 * w0.z + a1 * w1.z + a2 * w2.z; v = v > 0.f ? v : 0.2f * v; s += v * at.z;
      v = l.w + r.w + a0 * w0.w + a1 * w1.w + a2 * w2.w; v = v > 0.f ? v : 0.2f * v; s += v * at.w;
    }
  } else {
    for (int f = 0; f < F; ++f) {
      float v = xl[(size_t)si * F + f] + xr[(size_t)di * F + f]
              + a0 * We[f] + a1 * We[F + f] + a2 * We[2 * F + f];
      v = v > 0.0f ? v : 0.2f * v;
      s += v * att[f];
    }
  }
  if (!ok) s = -INFINITY;
  sout[e] = s;
  if (ok) atomicMax(menc + di, ord_enc(s));
}
__global__ void k_edge_exp(const float* __restrict__ s, const int* __restrict__ dst,
                           const unsigned char* __restrict__ mask,
                           const unsigned int* __restrict__ menc,
                           float* __restrict__ al, float* __restrict__ denom, int E, int n) {
  int e = blockIdx.x * blockDim.x + threadIdx.x;
  if (e >= E + n) return;
  int di; bool ok;
  if (e < E) { di = dst[e]; ok = mask ? (mask[e] != 0) : true; }
  else       { di = e - E;  ok = true; }
  float v = 0.0f;
  if (ok) v = expf(s[e] - ord_dec(menc[di]));
  al[e] = v;
  if (v != 0.0f) atomicAdd(denom + di, v);
}
__global__ void k_init_bias(float* out, const float* b, int n, int F) {
  int i = blockIdx.x * blockDim.x + threadIdx.x;
  if (i < n * F) out[i] = b[i % F];
}
__global__ void k_edge_aggr(const float* __restrict__ al, const float* __restrict__ denom,
                            const int* __restrict__ src, const int* __restrict__ dst,
                            const float* __restrict__ xl, float* __restrict__ out,
                            int E, int n, int F) {
  int e = blockIdx.x * blockDim.x + threadIdx.x;
  if (e >= E + n) return;
  int si, di;
  if (e < E) { si = src[e]; di = dst[e]; } else { si = di = e - E; }
  float w = al[e];
  if (w == 0.0f) return;
  w /= denom[di];
  if ((F & 3) == 0) {
    const float4* l4 = (const float4*)(xl + (size_t)si * F);
    float* o = out + (size_t)di * F;
    for (int f = 0; f < F; f += 4) {
      float4 l = l4[f >> 2];
      atomicAdd(o + f + 0, w * l.x);
      atomicAdd(o + f + 1, w * l.y);
      atomicAdd(o + f + 2, w * l.z);
      atomicAdd(o + f + 3, w * l.w);
    }
  } else {
    for (int f = 0; f < F; ++f) atomicAdd(out + (size_t)di * F + f, w * xl[(size_t)si * F + f]);
  }
}

// ---------- top-k via bitonic sort (descending score, idx tie-break) ----------
__global__ void k_sort_init(const float* score, float* key, int* idx, int n, int NP) {
  int i = blockIdx.x * blockDim.x + threadIdx.x;
  if (i >= NP) return;
  key[i] = (i < n) ? score[i] : -INFINITY;
  idx[i] = (i < n) ? i : 0x7FFFFFFF;
}
__global__ void k_bitonic(float* key, int* idx, int NP, int k, int j) {
  int i = blockIdx.x * blockDim.x + threadIdx.x;
  if (i >= NP) return;
  int l = i ^ j;
  if (l <= i) return;
  float ki = key[i], kl = key[l];
  int ii = idx[i], il = idx[l];
  bool l_before_i = (kl > ki) || (kl == ki && il < ii);
  bool up = ((i & k) == 0);
  if (up ? l_before_i : !l_before_i) {
    key[i] = kl; key[l] = ki; idx[i] = il; idx[l] = ii;
  }
}
__global__ void k_topk_commit(const float* key, const int* idx, float* tops, int* perm,
                              unsigned char* kept, int* inv, int kc) {
  int i = blockIdx.x * blockDim.x + threadIdx.x;
  if (i >= kc) return;
  int p = idx[i];
  perm[i] = p; tops[i] = key[i];
  kept[p] = 1; inv[p] = i;
}
__global__ void k_gather_scale(const float* x, const int* perm, const float* tops,
                               float* xp, int kc, int F) {
  int i = blockIdx.x * blockDim.x + threadIdx.x;
  if (i >= kc * F) return;
  int row = i / F, c = i - row * F;
  xp[i] = x[(size_t)perm[row] * F + c] * tops[row];
}
__global__ void k_gather3(const float* pos, const int* perm, float* posp, int kc) {
  int i = blockIdx.x * blockDim.x + threadIdx.x;
  if (i >= kc * 3) return;
  int row = i / 3, c = i - row * 3;
  posp[i] = pos[perm[row] * 3 + c];
}
__global__ void k_pool_edges(const int* src, const int* dst, const unsigned char* kept,
                             const int* inv, int* ps, int* pd, unsigned char* pm, int E) {
  int e = blockIdx.x * blockDim.x + threadIdx.x;
  if (e >= E) return;
  int s = src[e], d = dst[e];
  bool m = kept[s] && kept[d];
  pm[e] = m;
  ps[e] = m ? inv[s] : 0;
  pd[e] = m ? inv[d] : 0;
}

// ---------- kNN interpolate (k=3) ----------
__global__ void k_knn(const float* __restrict__ xs, const float* __restrict__ px,
                      const float* __restrict__ py, float* __restrict__ out,
                      int nx, int ny, int F) {
  int i = blockIdx.x * blockDim.x + threadIdx.x;
  if (i >= ny) return;
  float p0 = py[i * 3], p1 = py[i * 3 + 1], p2 = py[i * 3 + 2];
  float b0 = INFINITY, b1 = INFINITY, b2 = INFINITY;
  int i0 = 0, i1 = 0, i2 = 0;
  for (int j = 0; j < nx; ++j) {
    if ((j & 31) == 0 && j + 32 < nx) __builtin_prefetch(px + (j + 32) * 3, 0, 1);
    float dx = p0 - px[j * 3], dy = p1 - px[j * 3 + 1], dz = p2 - px[j * 3 + 2];
    float d2 = dx * dx + dy * dy + dz * dz;
    if (d2 < b0)      { b2 = b1; i2 = i1; b1 = b0; i1 = i0; b0 = d2; i0 = j; }
    else if (d2 < b1) { b2 = b1; i2 = i1; b1 = d2; i1 = j; }
    else if (d2 < b2) { b2 = d2; i2 = j; }
  }
  float w0 = 1.0f / fmaxf(b0, 1e-16f);
  float w1 = 1.0f / fmaxf(b1, 1e-16f);
  float w2 = 1.0f / fmaxf(b2, 1e-16f);
  float ws = w0 + w1 + w2;
  for (int f = 0; f < F; ++f)
    out[(size_t)i * F + f] = (w0 * xs[(size_t)i0 * F + f] + w1 * xs[(size_t)i1 * F + f]
                            + w2 * xs[(size_t)i2 * F + f]) / ws;
}

// =====================================================================
// host side
// =====================================================================
struct Gat { const float *Wl, *bl, *Wr, *br, *We, *att, *b; int fin, fout; };

static inline int gsz(int n) { return (n + TPB - 1) / TPB; }

static void run_gatv2(const float* x, int n, int fin, int fout,
                      const int* src, const int* dst, int E,
                      const float* ea, const unsigned char* mask,
                      const Gat& P, float* out,
                      _Float16* X16, _Float16* W16a, _Float16* W16b,
                      float* xl, float* xr, float* slacc, float* slattr,
                      float* sbuf, float* albuf, unsigned int* menc, float* denom,
                      hipStream_t stream) {
  const int finP = (fin + 31) & ~31;
  const int foutP = (fout + 15) & ~15;
  const int nP = (n + 15) & ~15;
  k_cvt_pad<<<gsz(nP * finP), TPB, 0, stream>>>(x, X16, n, fin, nP, finP);
  k_cvt_pad<<<gsz(finP * foutP), TPB, 0, stream>>>(P.Wl, W16a, fin, fout, finP, foutP);
  k_cvt_pad<<<gsz(finP * foutP), TPB, 0, stream>>>(P.Wr, W16b, fin, fout, finP, foutP);
  dim3 gg((nP / 16 + 7) / 8, foutP / 16);
  k_gemm_f16<<<gg, 256, 0, stream>>>(X16, W16a, P.bl, xl, n, fout, finP, foutP);
  k_gemm_f16<<<gg, 256, 0, stream>>>(X16, W16b, P.br, xr, n, fout, finP, foutP);
  k_fill_u32<<<gsz(n * 4), TPB, 0, stream>>>((unsigned int*)slacc, 0u, n * 4);
  k_slattr_accum<<<gsz(E), TPB, 0, stream>>>(dst, ea, mask, slacc, E);
  k_slattr_norm<<<gsz(n), TPB, 0, stream>>>(slacc, slattr, n);
  k_fill_u32<<<gsz(n), TPB, 0, stream>>>(menc, 0u, n);
  k_fill_u32<<<gsz(n), TPB, 0, stream>>>((unsigned int*)denom, 0u, n);
  int T = E + n;
  k_edge_score<<<gsz(T), TPB, 0, stream>>>(src, dst, ea, slattr, mask, xl, xr,
                                           P.We, P.att, sbuf, menc, E, n, fout);
  k_edge_exp<<<gsz(T), TPB, 0, stream>>>(sbuf, dst, mask, menc, albuf, denom, E, n);
  k_init_bias<<<gsz(n * fout), TPB, 0, stream>>>(out, P.b, n, fout);
  k_edge_aggr<<<gsz(T), TPB, 0, stream>>>(albuf, denom, src, dst, xl, out, E, n, fout);
}

static void run_topk_sort(const float* score, int n, int NP, float* key, int* idx,
                          float* tops, int* perm, unsigned char* kept, int* inv,
                          int kc, hipStream_t stream) {
  k_sort_init<<<gsz(NP), TPB, 0, stream>>>(score, key, idx, n, NP);
  for (int k = 2; k <= NP; k <<= 1)
    for (int j = k >> 1; j > 0; j >>= 1)
      k_bitonic<<<gsz(NP), TPB, 0, stream>>>(key, idx, NP, k, j);
  k_fill_u8<<<gsz(n), TPB, 0, stream>>>(kept, 0, n);
  k_fill_u32<<<gsz(n), TPB, 0, stream>>>((unsigned int*)inv, 0u, n);
  k_topk_commit<<<gsz(kc), TPB, 0, stream>>>(key, idx, tops, perm, kept, inv, kc);
}

extern "C" void kernel_launch(void* const* d_in, const int* in_sizes, int n_in,
                              void* d_out, int out_size, void* d_ws, size_t ws_size,
                              hipStream_t stream) {
  const int N3 = 16000, N2 = 16000, H = 128, L = 64, F3 = 5;
  const int KP = 4000, NP = 16384;

  const float* x2   = (const float*)d_in[3];
  const float* pos3 = (const float*)d_in[2];
  const float* pos2 = (const float*)d_in[5];
  const int* ei3  = (const int*)d_in[1]; const int E3  = in_sizes[1] / 2;
  const int* ei2  = (const int*)d_in[4]; const int E2  = in_sizes[4] / 2;
  const int* eia3 = (const int*)d_in[6]; const int EA3 = in_sizes[6] / 2;
  const int* eia2 = (const int*)d_in[7]; const int EA2 = in_sizes[7] / 2;
  const int* s3 = ei3, *d3 = ei3 + E3;
  const int* s2 = ei2, *d2 = ei2 + E2;
  const int* sa3 = eia3, *da3 = eia3 + EA3;
  const int* sa2 = eia2, *da2 = eia2 + EA2;

  // params in dict insertion order s0,s1,sp,e0,e1,ep,e2,e3,d0,d1,d2,d3 × [Wl,bl,Wr,br,We,att,b]
  static const int fins[12]  = {4, 131, 131, 8, 131, 131, 131, 131, 67, 131, 131, 131};
  static const int fouts[12] = {H, H, 1, H, H, 1, H, L, H, H, H, F3};
  Gat G[12];
  int base = 8;
  for (int i = 0; i < 12; ++i) {
    G[i].Wl  = (const float*)d_in[base + 0];
    G[i].bl  = (const float*)d_in[base + 1];
    G[i].Wr  = (const float*)d_in[base + 2];
    G[i].br  = (const float*)d_in[base + 3];
    G[i].We  = (const float*)d_in[base + 4];
    G[i].att = (const float*)d_in[base + 5];
    G[i].b   = (const float*)d_in[base + 6];
    G[i].fin = fins[i]; G[i].fout = fouts[i];
    base += 7;
  }

  // bump allocator over workspace (byte-sized)
  size_t used = 0;
  auto alloc = [&](size_t bytes) -> void* {
    void* p = (char*)d_ws + used;
    used = (used + bytes + 255) & ~(size_t)255;
    return p;
  };
  int EMAX = E3; if (E2 > EMAX) EMAX = E2; if (EA3 > EMAX) EMAX = EA3; if (EA2 > EMAX) EMAX = EA2;
  EMAX += N3;

  const int FOUTP_MAX = 128, NP_MAX = 16016;
  _Float16* X16  = (_Float16*)alloc((size_t)NP_MAX * FINP_MAX * 2);
  _Float16* W16a = (_Float16*)alloc((size_t)FINP_MAX * FOUTP_MAX * 2);
  _Float16* W16b = (_Float16*)alloc((size_t)FINP_MAX * FOUTP_MAX * 2);
  float* xl    = (float*)alloc((size_t)N3 * H * 4);
  float* xr    = (float*)alloc((size_t)N3 * H * 4);
  float* cat   = (float*)alloc((size_t)N3 * (H + 3) * 4);
  float* slacc = (float*)alloc((size_t)N3 * 4 * 4);
  float* slat  = (float*)alloc((size_t)N3 * 3 * 4);
  unsigned int* menc = (unsigned int*)alloc((size_t)N3 * 4);
  float* denom = (float*)alloc((size_t)N3 * 4);
  float* sbuf  = (float*)alloc((size_t)EMAX * 4);
  float* albuf = (float*)alloc((size_t)EMAX * 4);
  float* h3    = (float*)alloc((size_t)N3 * H * 4);
  float* z2    = (float*)alloc((size_t)N2 * H * 4);
  float* ea3   = (float*)alloc((size_t)E3 * 3 * 4);
  float* ea2   = (float*)alloc((size_t)E2 * 3 * 4);
  float* eaA3  = (float*)alloc((size_t)EA3 * 3 * 4);
  float* eaA2  = (float*)alloc((size_t)EA2 * 3 * 4);
  float* deg   = (float*)alloc((size_t)N3 * 4);
  float* sc1   = (float*)alloc((size_t)N3 * 4);
  float* score = (float*)alloc((size_t)N3 * 4);
  float* skey  = (float*)alloc((size_t)NP * 4);
  int*   sidx  = (int*)alloc((size_t)NP * 4);
  float* tops3 = (float*)alloc((size_t)KP * 4); int* perm3 = (int*)alloc((size_t)KP * 4);
  unsigned char* kept3 = (unsigned char*)alloc(N3); int* inv3 = (int*)alloc((size_t)N3 * 4);
  float* tops2 = (float*)alloc((size_t)KP * 4); int* perm2 = (int*)alloc((size_t)KP * 4);
  unsigned char* kept2 = (unsigned char*)alloc(N2); int* inv2 = (int*)alloc((size_t)N2 * 4);
  int* ps3 = (int*)alloc((size_t)E3 * 4); int* pd3 = (int*)alloc((size_t)E3 * 4);
  unsigned char* pm3 = (unsigned char*)alloc(E3);
  int* ps2 = (int*)alloc((size_t)E2 * 4); int* pd2 = (int*)alloc((size_t)E2 * 4);
  unsigned char* pm2 = (unsigned char*)alloc(E2);
  float* pos2p = (float*)alloc((size_t)KP * 3 * 4);
  float* zp    = (float*)alloc((size_t)KP * H * 4);
  float* lat   = (float*)alloc((size_t)KP * L * 4);
  float* yk    = (float*)alloc((size_t)KP * H * 4);
  float* yi    = (float*)alloc((size_t)N3 * H * 4);
  float* eap   = (float*)alloc((size_t)E3 * 3 * 4);
  float* ea2p  = (float*)alloc((size_t)E2 * 3 * 4);
  (void)ws_size; (void)n_in; (void)out_size;

  // ---- edge attributes + degree ----
  k_eattr<<<gsz(E3), TPB, 0, stream>>>(s3, d3, pos3, ea3, E3);
  k_eattr<<<gsz(E2), TPB, 0, stream>>>(s2, d2, pos2, ea2, E2);
  k_eattr<<<gsz(EA3), TPB, 0, stream>>>(sa3, da3, pos3, eaA3, EA3);
  k_eattr<<<gsz(EA2), TPB, 0, stream>>>(sa2, da2, pos2, eaA2, EA2);
  k_fill_u32<<<gsz(N3), TPB, 0, stream>>>((unsigned int*)deg, 0u, N3);
  k_deg<<<gsz(E3), TPB, 0, stream>>>(s3, deg, E3);

  // ---- StructureEncoder (3D) ----
  k_concat_pos<<<gsz(N3 * 4), TPB, 0, stream>>>(deg, pos3, cat, N3, 1);
  run_gatv2(cat, N3, 4, H, s3, d3, E3, ea3, nullptr, G[0], h3,
            X16, W16a, W16b, xl, xr, slacc, slat, sbuf, albuf, menc, denom, stream);
  k_elu<<<gsz(N3 * H), TPB, 0, stream>>>(h3, N3 * H);
  k_concat_pos<<<gsz(N3 * 131), TPB, 0, stream>>>(h3, pos3, cat, N3, H);
  run_gatv2(cat, N3, 131, H, s3, d3, E3, ea3, nullptr, G[1], h3,
            X16, W16a, W16b, xl, xr, slacc, slat, sbuf, albuf, menc, denom, stream);
  k_elu<<<gsz(N3 * H), TPB, 0, stream>>>(h3, N3 * H);
  // sag_pool 3D (x_p / pos3_p unused downstream)
  k_concat_pos<<<gsz(N3 * 131), TPB, 0, stream>>>(h3, pos3, cat, N3, H);
  run_gatv2(cat, N3, 131, 1, sa3, da3, EA3, eaA3, nullptr, G[2], sc1,
            X16, W16a, W16b, xl, xr, slacc, slat, sbuf, albuf, menc, denom, stream);
  k_tanh<<<gsz(N3), TPB, 0, stream>>>(sc1, score, N3);
  run_topk_sort(score, N3, NP, skey, sidx, tops3, perm3, kept3, inv3, KP, stream);
  k_pool_edges<<<gsz(E3), TPB, 0, stream>>>(s3, d3, kept3, inv3, ps3, pd3, pm3, E3);

  // ---- Encoder (2D) ----
  k_concat_pos<<<gsz(N2 * 8), TPB, 0, stream>>>(x2, pos2, cat, N2, 5);
  run_gatv2(cat, N2, 8, H, s2, d2, E2, ea2, nullptr, G[3], z2,
            X16, W16a, W16b, xl, xr, slacc, slat, sbuf, albuf, menc, denom, stream);
  k_elu<<<gsz(N2 * H), TPB, 0, stream>>>(z2, N2 * H);
  k_concat_pos<<<gsz(N2 * 131), TPB, 0, stream>>>(z2, pos2, cat, N2, H);
  run_gatv2(cat, N2, 131, H, s2, d2, E2, ea2, nullptr, G[4], z2,
            X16, W16a, W16b, xl, xr, slacc, slat, sbuf, albuf, menc, denom, stream);
  k_elu<<<gsz(N2 * H), TPB, 0, stream>>>(z2, N2 * H);
  // sag_pool 2D
  k_concat_pos<<<gsz(N2 * 131), TPB, 0, stream>>>(z2, pos2, cat, N2, H);
  run_gatv2(cat, N2, 131, 1, sa2, da2, EA2, eaA2, nullptr, G[5], sc1,
            X16, W16a, W16b, xl, xr, slacc, slat, sbuf, albuf, menc, denom, stream);
  k_tanh<<<gsz(N2), TPB, 0, stream>>>(sc1, score, N2);
  run_topk_sort(score, N2, NP, skey, sidx, tops2, perm2, kept2, inv2, KP, stream);
  k_pool_edges<<<gsz(E2), TPB, 0, stream>>>(s2, d2, kept2, inv2, ps2, pd2, pm2, E2);
  k_gather3<<<gsz(KP * 3), TPB, 0, stream>>>(pos2, perm2, pos2p, KP);
  k_gather_scale<<<gsz(KP * H), TPB, 0, stream>>>(z2, perm2, tops2, zp, KP, H);
  // pooled 2D convs
  k_eattr<<<gsz(E2), TPB, 0, stream>>>(ps2, pd2, pos2p, ea2p, E2);
  k_concat_pos<<<gsz(KP * 131), TPB, 0, stream>>>(zp, pos2p, cat, KP, H);
  run_gatv2(cat, KP, 131, H, ps2, pd2, E2, ea2p, pm2, G[6], yk,
            X16, W16a, W16b, xl, xr, slacc, slat, sbuf, albuf, menc, denom, stream);
  k_elu<<<gsz(KP * H), TPB, 0, stream>>>(yk, KP * H);
  k_concat_pos<<<gsz(KP * 131), TPB, 0, stream>>>(yk, pos2p, cat, KP, H);
  run_gatv2(cat, KP, 131, L, ps2, pd2, E2, ea2p, pm2, G[7], lat,
            X16, W16a, W16b, xl, xr, slacc, slat, sbuf, albuf, menc, denom, stream);
  k_elu<<<gsz(KP * L), TPB, 0, stream>>>(lat, KP * L);

  // ---- DBA: pooled-3D edges with pooled-2D positions ----
  k_eattr<<<gsz(E3), TPB, 0, stream>>>(ps3, pd3, pos2p, eap, E3);
  k_concat_pos<<<gsz(KP * 67), TPB, 0, stream>>>(lat, pos2p, cat, KP, L);
  run_gatv2(cat, KP, 67, H, ps3, pd3, E3, eap, pm3, G[8], yk,
            X16, W16a, W16b, xl, xr, slacc, slat, sbuf, albuf, menc, denom, stream);
  k_elu<<<gsz(KP * H), TPB, 0, stream>>>(yk, KP * H);
  k_concat_pos<<<gsz(KP * 131), TPB, 0, stream>>>(yk, pos2p, cat, KP, H);
  run_gatv2(cat, KP, 131, H, ps3, pd3, E3, eap, pm3, G[9], yk,
            X16, W16a, W16b, xl, xr, slacc, slat, sbuf, albuf, menc, denom, stream);
  k_elu<<<gsz(KP * H), TPB, 0, stream>>>(yk, KP * H);

  // ---- kNN interpolate up to full 3D graph ----
  k_knn<<<gsz(N3), TPB, 0, stream>>>(yk, pos2p, pos3, yi, KP, N3, H);

  // ---- decoder tail on full 3D graph ----
  k_concat_pos<<<gsz(N3 * 131), TPB, 0, stream>>>(yi, pos3, cat, N3, H);
  run_gatv2(cat, N3, 131, H, s3, d3, E3, ea3, nullptr, G[10], yi,
            X16, W16a, W16b, xl, xr, slacc, slat, sbuf, albuf, menc, denom, stream);
  k_elu<<<gsz(N3 * H), TPB, 0, stream>>>(yi, N3 * H);
  k_concat_pos<<<gsz(N3 * 131), TPB, 0, stream>>>(yi, pos3, cat, N3, H);
  run_gatv2(cat, N3, 131, F3, s3, d3, E3, ea3, nullptr, G[11], (float*)d_out,
            X16, W16a, W16b, xl, xr, slacc, slat, sbuf, albuf, menc, denom, stream);
}
